// GATv2_22539988370024
// MI455X (gfx1250) — matrix operand order repaired and checked by reading, not compiled, not run on hardware
//
#include <hip/hip_runtime.h>

#define HD   512      // hidden / feature dim (all layers)
#define NSEG 8192     // B*N nodes
#define NG   1024     // nodes per graph
#define NGR  8        // graphs

typedef __attribute__((ext_vector_type(16))) _Float16 v16h;
typedef __attribute__((ext_vector_type(8)))  float    v8f;

// ---------- helpers ----------
// order-preserving float<->uint key for atomicMax on signed floats
__device__ __forceinline__ unsigned fkey(float f) {
  unsigned u = __float_as_uint(f);
  return (u & 0x80000000u) ? ~u : (u | 0x80000000u);
}
__device__ __forceinline__ float fdecode(unsigned k) {
  unsigned u = (k & 0x80000000u) ? (k & 0x7FFFFFFFu) : ~k;
  return __uint_as_float(u);
}

// async global->LDS 16B copy (ASYNCcnt path, gfx1250)
__device__ __forceinline__ void async_copy_b128(unsigned lds_addr, const void* gaddr) {
  unsigned long long ga = (unsigned long long)(size_t)gaddr;
  asm volatile("global_load_async_to_lds_b128 %0, %1, off"
               :: "v"(lds_addr), "v"(ga) : "memory");
}
__device__ __forceinline__ void wait_async_zero() {
  asm volatile("s_wait_asynccnt 0x0" ::: "memory");
}

// ---------- conversion / packing ----------
__global__ void f32_to_f16_kernel(const float* __restrict__ in,
                                  _Float16* __restrict__ out, int n) {
  int i = blockIdx.x * blockDim.x + threadIdx.x;
  if (i < n) out[i] = (_Float16)in[i];
}

// Pack W [K=512, N=512] row-major f32 -> f16 WMMA B-fragment order.
// Packed index = ((nt*16 + kt)*32 + lane)*16 + i
// Packed B (contiguous k, as A): lanes 0-15 hold K=0..15, lanes 16-31 hold K=16..31; col = nt*16 + (lane&15)
__global__ void pack_w_kernel(const float* __restrict__ W,
                              _Float16* __restrict__ Wp) {
  int idx  = blockIdx.x * blockDim.x + threadIdx.x;   // < 512*512
  int i    = idx & 15;
  int lane = (idx >> 4) & 31;
  int kt   = (idx >> 9) & 15;
  int nt   = idx >> 13;
  int half = lane >> 4;
  int k = kt * 32 + half * 16 + i;
  int n = nt * 16 + (lane & 15);
  Wp[idx] = (_Float16)W[k * HD + n];
}

union Frag { v16h v; uint4 q[2]; };

// ---------- feature GEMM: out[8192,512](f32) = A(f16[8192,512]) x Wpacked + bias ----------
// A tile staged in LDS once per block via async copy; B streams from global/L2.
__global__ __launch_bounds__(256) void gemm_feat_kernel(
    const _Float16* __restrict__ A,
    const _Float16* __restrict__ Wp,
    const float* __restrict__ bias,
    float* __restrict__ out) {
  __shared__ _Float16 atile[16 * HD];          // 16 KB: this block's 16 A-rows
  const int tid   = threadIdx.x;
  const int wave  = tid >> 5;
  const int lane  = tid & 31;
  const int mtile = blockIdx.x;                // all 8 waves share one M-tile
  const int ng    = wave;                      // 64-column group per wave
  const int half  = lane >> 4;

  // ---- async global->LDS stage of the A tile ----
  {
    const _Float16* gA = A + (size_t)mtile * 16 * HD;
    const unsigned lbase = (unsigned)(size_t)(&atile[0]);
#pragma unroll
    for (int j = 0; j < 4; ++j) {
      const int c = tid + 256 * j;             // 1024 x 16-byte chunks
      async_copy_b128(lbase + c * 16, gA + c * 8);
    }
    wait_async_zero();
  }
  __syncthreads();

  const _Float16* arow = atile + (lane & 15) * HD;
  const _Float16* wl   = Wp + (size_t)(ng * 4) * 8192 + lane * 16;

  v8f c[4] = {{}, {}, {}, {}};
  Frag a, b[2][4];
#pragma unroll
  for (int t = 0; t < 4; ++t) {                // preload B for kt = 0
    b[0][t].q[0] = *(const uint4*)(wl + t * 8192);
    b[0][t].q[1] = *(const uint4*)(wl + t * 8192 + 8);
  }
#pragma unroll
  for (int kt = 0; kt < 16; ++kt) {
    const int cur = kt & 1, nxt = cur ^ 1;
    const int kb = kt * 32 + half * 16;
    a.q[0] = *(const uint4*)(arow + kb);       // ds_load_b128 from LDS
    a.q[1] = *(const uint4*)(arow + kb + 8);
    if (kt < 15) {                             // prefetch next B K-slice
      const _Float16* bp = wl + (kt + 1) * 512;
#pragma unroll
      for (int t = 0; t < 4; ++t) {
        b[nxt][t].q[0] = *(const uint4*)(bp + t * 8192);
        b[nxt][t].q[1] = *(const uint4*)(bp + t * 8192 + 8);
      }
    }
#pragma unroll
    for (int t = 0; t < 4; ++t)
      c[t] = __builtin_amdgcn_wmma_f32_16x16x32_f16(
          false, a.v, false, b[cur][t].v, (short)0, c[t], false, false);
  }
#pragma unroll
  for (int t = 0; t < 4; ++t) {
    const int col = (ng * 4 + t) * 16 + (lane & 15);
    const float bv = bias[col];
#pragma unroll
    for (int r = 0; r < 8; ++r)
      out[(size_t)(mtile * 16 + half * 8 + r) * HD + col] = c[t][r] + bv;
  }
}

// ---------- LDS-tiled transpose+convert: xl f32[8192,512] -> xlT f16[512,8192] ----------
__global__ __launch_bounds__(256) void transpose_f16_kernel(
    const float* __restrict__ in, _Float16* __restrict__ out) {
  __shared__ _Float16 tile[32][33];
  const int hb = blockIdx.x * 32;
  const int nb = blockIdx.y * 32;
  const int x = threadIdx.x;
#pragma unroll
  for (int r = threadIdx.y; r < 32; r += 8)
    tile[r][x] = (_Float16)in[(size_t)(nb + r) * HD + hb + x];
  __syncthreads();
#pragma unroll
  for (int r = threadIdx.y; r < 32; r += 8)
    out[(size_t)(hb + r) * NSEG + nb + x] = tile[x][r];
}

// ---------- segment softmax ----------
__global__ void seg_init_kernel(unsigned* __restrict__ mkey, float* __restrict__ s, int n) {
  int i = blockIdx.x * blockDim.x + threadIdx.x;
  if (i < n) { mkey[i] = 0x007FFFFFu; /* fkey(-inf) */ s[i] = 0.f; }
}

// one wave per edge: e = a . leaky_relu(xl[src] + xr[dst]); atomicMax per dst
__global__ __launch_bounds__(256) void edge_score_kernel(
    const float* __restrict__ xl, const float* __restrict__ xr,
    const float* __restrict__ avec, const int* __restrict__ src,
    const int* __restrict__ dst, float* __restrict__ e,
    unsigned* __restrict__ mkey, int E) {
  const int lane = threadIdx.x & 31;
  const int edge = blockIdx.x * 8 + (threadIdx.x >> 5);
  if (edge >= E) return;
  const int sj = src[edge], di = dst[edge];
  const float* pl = xl + (size_t)sj * HD;
  const float* pr = xr + (size_t)di * HD;
  float acc = 0.f;
#pragma unroll
  for (int t = 0; t < HD / 32; ++t) {
    const int h = lane + t * 32;
    float v = pl[h] + pr[h];
    v = v > 0.f ? v : 0.2f * v;
    acc += v * avec[h];
  }
  for (int off = 16; off; off >>= 1) acc += __shfl_xor(acc, off, 32);
  if (lane == 0) {
    e[edge] = acc;
    atomicMax(mkey + di, fkey(acc));
  }
}

__global__ void edge_exp_kernel(float* __restrict__ e, const unsigned* __restrict__ mkey,
                                const int* __restrict__ dst, float* __restrict__ s, int E) {
  int i = blockIdx.x * blockDim.x + threadIdx.x;
  if (i >= E) return;
  const int di = dst[i];
  const float ex = __expf(e[i] - fdecode(mkey[di]));
  e[i] = ex;
  atomicAdd(s + di, ex);
}

__global__ void zero_kernel(uint4* __restrict__ p, int n4) {
  int i = blockIdx.x * blockDim.x + threadIdx.x;
  if (i < n4) p[i] = uint4{0u, 0u, 0u, 0u};
}

// scatter normalized attention into dense per-graph alpha matrix [8][1024][1024] f16
__global__ void edge_alpha_kernel(const float* __restrict__ e, const float* __restrict__ s,
                                  const int* __restrict__ src, const int* __restrict__ dst,
                                  _Float16* __restrict__ alpha, int E) {
  int i = blockIdx.x * blockDim.x + threadIdx.x;
  if (i >= E) return;
  const int di = dst[i], sj = src[i];
  const int g = di >> 10;
  alpha[((size_t)g * NG + (di & (NG - 1))) * NG + (sj & (NG - 1))] =
      (_Float16)(e[i] / s[di]);
}

// ---------- aggregation GEMM: out[g*1024+i, h] = sum_j alpha[g][i][j] * xlT[h][g*1024+j] + bias[h]
template <typename OutT>
__global__ __launch_bounds__(256) void gemm_aggr_kernel(
    const _Float16* __restrict__ alpha,   // [8][1024][1024] row-major
    const _Float16* __restrict__ xlT,     // [512][8192]
    const float* __restrict__ bias,
    OutT* __restrict__ out) {             // [8192][512]
  __shared__ _Float16 atile[16 * NG];     // 32 KB: this block's 16 alpha rows
  const int tid  = threadIdx.x;
  const int wave = tid >> 5;
  const int lane = tid & 31;
  const int ng = wave;                    // 64-col group of H
  const int mt = blockIdx.x & 63;         // M-tile within graph (shared by block)
  const int g  = blockIdx.x >> 6;         // graph (shared by block)
  const int half = lane >> 4;

  // ---- async global->LDS stage of the alpha tile ----
  {
    const _Float16* gA = alpha + ((size_t)g * NG + mt * 16) * NG;
    const unsigned lbase = (unsigned)(size_t)(&atile[0]);
#pragma unroll
    for (int j = 0; j < 8; ++j) {
      const int c = tid + 256 * j;        // 2048 x 16-byte chunks
      async_copy_b128(lbase + c * 16, gA + c * 8);
    }
    wait_async_zero();
  }
  __syncthreads();

  const _Float16* arow = atile + (lane & 15) * NG;

  v8f c[4] = {{}, {}, {}, {}};
  Frag a, b[2][4];
  int cols[4];
  const _Float16* bp[4];
#pragma unroll
  for (int t = 0; t < 4; ++t) {
    cols[t] = (ng * 4 + t) * 16 + (lane & 15);
    bp[t] = xlT + (size_t)cols[t] * NSEG + g * NG + half * 16;
  }
#pragma unroll
  for (int t = 0; t < 4; ++t) {           // preload B for kt = 0
    b[0][t].q[0] = *(const uint4*)(bp[t]);
    b[0][t].q[1] = *(const uint4*)(bp[t] + 8);
  }
#pragma unroll 4
  for (int kt = 0; kt < 32; ++kt) {
    const int cur = kt & 1, nxt = cur ^ 1;
    const int kb = kt * 32 + half * 16;
    a.q[0] = *(const uint4*)(arow + kb);  // ds_load_b128
    a.q[1] = *(const uint4*)(arow + kb + 8);
    if (kt < 31) {
      const int kn = (kt + 1) * 32;
#pragma unroll
      for (int t = 0; t < 4; ++t) {
        b[nxt][t].q[0] = *(const uint4*)(bp[t] + kn);
        b[nxt][t].q[1] = *(const uint4*)(bp[t] + kn + 8);
      }
    }
#pragma unroll
    for (int t = 0; t < 4; ++t)
      c[t] = __builtin_amdgcn_wmma_f32_16x16x32_f16(
          false, a.v, false, b[cur][t].v, (short)0, c[t], false, false);
  }
#pragma unroll
  for (int t = 0; t < 4; ++t) {
    const float bv = bias[cols[t]];
#pragma unroll
    for (int r = 0; r < 8; ++r) {
      const int node = g * NG + mt * 16 + half * 8 + r;
      out[(size_t)node * HD + cols[t]] = (OutT)(c[t][r] + bv);
    }
  }
}

// ---------- driver ----------
extern "C" void kernel_launch(void* const* d_in, const int* in_sizes, int n_in,
                              void* d_out, int out_size, void* d_ws, size_t ws_size,
                              hipStream_t stream) {
  const float* x0 = (const float*)d_in[0];
  const int*   ei = (const int*)d_in[2];
  const int E = in_sizes[2] / 2;
  const int* src = ei;
  const int* dst = ei + E;

  char* ws = (char*)d_ws;
  auto alloc = [&](size_t bytes) {
    char* p = ws;
    ws += (bytes + 255) & ~(size_t)255;
    return p;
  };
  _Float16* xh  = (_Float16*)alloc((size_t)NSEG * HD * 2);   // layer input, f16
  _Float16* wlp = (_Float16*)alloc((size_t)HD * HD * 2);
  _Float16* wrp = (_Float16*)alloc((size_t)HD * HD * 2);
  float* xl     = (float*)alloc((size_t)NSEG * HD * 4);
  char*  xr_al  = alloc((size_t)NSEG * HD * 4);              // xr f32, later aliased by alpha f16
  float* xr     = (float*)xr_al;
  _Float16* alpha = (_Float16*)xr_al;                        // [8][1024][1024] f16, 16 MB
  _Float16* xlT = (_Float16*)alloc((size_t)HD * NSEG * 2);
  float* ebuf   = (float*)alloc((size_t)E * 4);
  unsigned* mkey = (unsigned*)alloc((size_t)NSEG * 4);
  float* sbuf   = (float*)alloc((size_t)NSEG * 4);

  const int nElem = NSEG * HD;
  f32_to_f16_kernel<<<nElem / 256, 256, 0, stream>>>(x0, xh, nElem);

  for (int L = 0; L < 3; ++L) {
    const float* Wl   = (const float*)d_in[3 + L * 6 + 0];
    const float* bl   = (const float*)d_in[3 + L * 6 + 1];
    const float* Wr   = (const float*)d_in[3 + L * 6 + 2];
    const float* br   = (const float*)d_in[3 + L * 6 + 3];
    const float* av   = (const float*)d_in[3 + L * 6 + 4];
    const float* bias = (const float*)d_in[3 + L * 6 + 5];

    pack_w_kernel<<<(HD * HD) / 256, 256, 0, stream>>>(Wl, wlp);
    pack_w_kernel<<<(HD * HD) / 256, 256, 0, stream>>>(Wr, wrp);
    gemm_feat_kernel<<<512, 256, 0, stream>>>(xh, wlp, bl, xl);
    gemm_feat_kernel<<<512, 256, 0, stream>>>(xh, wrp, br, xr);
    transpose_f16_kernel<<<dim3(HD / 32, NSEG / 32), dim3(32, 8), 0, stream>>>(xl, xlT);
    seg_init_kernel<<<NSEG / 256, 256, 0, stream>>>(mkey, sbuf, NSEG);
    edge_score_kernel<<<(E + 7) / 8, 256, 0, stream>>>(xl, xr, av, src, dst, ebuf, mkey, E);
    edge_exp_kernel<<<(E + 255) / 256, 256, 0, stream>>>(ebuf, mkey, dst, sbuf, E);
    // xr is dead from here; reuse its buffer as the dense alpha matrix
    zero_kernel<<<(NGR * NG * NG * 2 / 16) / 256, 256, 0, stream>>>((uint4*)alpha,
                                                                    NGR * NG * NG * 2 / 16);
    edge_alpha_kernel<<<(E + 255) / 256, 256, 0, stream>>>(ebuf, sbuf, src, dst, alpha, E);
    if (L < 2) {
      gemm_aggr_kernel<_Float16><<<512, 256, 0, stream>>>(alpha, xlT, bias, xh);
    } else {
      gemm_aggr_kernel<float><<<512, 256, 0, stream>>>(alpha, xlT, bias, (float*)d_out);
    }
  }
}
